// C2QAttention_34007551049865
// MI455X (gfx1250) — matrix operand order repaired and checked by reading, not compiled, not run on hardware
//
#include <hip/hip_runtime.h>
#include <hip/hip_bf16.h>
#include <math.h>

// C2Q attention: out[b,c,d] = sum_q softmax_q(S[b,c,q]) * Qenc[b,q,d]
// B=32, C=2048, Q=512, D=1024, all fp32 in/out.
//
// MI455X (gfx1250, wave32) strategy:
//   * 128x128 fp32 output tile per 256-thread WG (8 waves, 2x4 grid; each
//     wave owns 64x32 = 4x2 v_wmma_f32_16x16x32_f16 accumulators).
//   * Online (single-sweep) softmax stats in fp32.
//   * Double-buffered LDS K-pipeline: global loads for tile kt+1 issue into
//     registers before tile kt's WMMAs, then drain into the other LDS buffer;
//     one barrier per K-step. __builtin_prefetch pulls tile kt+2 toward L2.
//   * A = exp(s-m)/sum in f16; B (Qenc) in f16 stored TRANSPOSED so each
//     lane's B fragment is a contiguous 32B LDS read.

typedef __attribute__((ext_vector_type(16))) _Float16 v16h;
typedef __attribute__((ext_vector_type(8)))  _Float16 v8h;
typedef __attribute__((ext_vector_type(4)))  _Float16 v4h;
typedef __attribute__((ext_vector_type(8)))  float    v8f;

#define TILE_C 128
#define TILE_D 128
#define TILE_K 32
#define LDA 40           // halfs per A row (32 used + 8 pad); 80B, 16B-aligned
#define LDB 40           // halfs per Bt row
#define ABUF (TILE_C * LDA)
#define BBUF (TILE_D * LDB)
#define C_DIM 2048
#define Q_DIM 512
#define D_DIM 1024
#define K_TILES (Q_DIM / TILE_K)   // 16

__global__ __launch_bounds__(256, 2)
void c2q_attn_wmma_kernel(const float* __restrict__ S,
                          const float* __restrict__ Qe,
                          float* __restrict__ out) {
    __shared__ __align__(16) _Float16 Alds[2 * ABUF];
    __shared__ __align__(16) _Float16 Blds[2 * BBUF];
    __shared__ float rowm[TILE_C];
    __shared__ float rowinv[TILE_C];
    __shared__ float pm[256];
    __shared__ float ps[256];

    const int tid  = threadIdx.x;
    const int lane = tid & 31;
    const int wave = tid >> 5;          // 0..7
    const int b    = blockIdx.y;        // batch
    const int ct   = blockIdx.x >> 3;   // C tile 0..15
    const int dt   = blockIdx.x & 7;    // D tile 0..7
    const int c0   = ct * TILE_C;
    const int d0   = dt * TILE_D;

    const float* Sb = S   + (size_t)b * C_DIM * Q_DIM;
    const float* Qb = Qe  + (size_t)b * Q_DIM * D_DIM;
    float*       Ob = out + (size_t)b * C_DIM * D_DIM;

    // ---------------- Phase 1: online softmax stats (single sweep) ---------
    {
        const int row = tid >> 1;       // 0..127
        const int seg = tid & 1;        // each half-row: 256 floats
        const float4* sp = (const float4*)(Sb + (size_t)(c0 + row) * Q_DIM + seg * 256);

        float m = -3.402823466e+38f;
        float s = 0.f;
        #pragma unroll 4
        for (int i = 0; i < 64; ++i) {
            float4 v = sp[i];
            float cm = fmaxf(fmaxf(v.x, v.y), fmaxf(v.z, v.w));
            float nm = fmaxf(m, cm);
            s = s * __expf(m - nm)
              + __expf(v.x - nm) + __expf(v.y - nm)
              + __expf(v.z - nm) + __expf(v.w - nm);
            m = nm;
        }
        pm[tid] = m;
        ps[tid] = s;
        __syncthreads();
        if (tid < TILE_C) {
            float m0 = pm[2 * tid], m1 = pm[2 * tid + 1];
            float nm = fmaxf(m0, m1);
            float ssum = ps[2 * tid] * __expf(m0 - nm)
                       + ps[2 * tid + 1] * __expf(m1 - nm);
            rowm[tid]   = nm;
            rowinv[tid] = 1.0f / ssum;
        }
        __syncthreads();
    }

    // ---------------- Phase 2: double-buffered K-loop WMMA GEMM ------------
    const int waveM = wave & 1;         // M offset 0/64
    const int waveN = wave >> 1;        // N offset 0/32/64/96
    const int mBase = waveM * 64;
    const int nBase = waveN * 32;
    const int hsel  = lane >> 4;        // lane half: 0 or 1
    const int lrow  = lane & 15;

    // Cooperative-load index maps (constant per thread).
    const int rA  = tid >> 1;           // unused; kept pattern below explicit
    (void)rA;

    v8f acc[4][2];
    #pragma unroll
    for (int m = 0; m < 4; ++m)
        #pragma unroll
        for (int n = 0; n < 2; ++n)
            acc[m][n] = (v8f){0.f, 0.f, 0.f, 0.f, 0.f, 0.f, 0.f, 0.f};

    float4 ra[4], rb[4];

    // -- global load of K-tile kt into registers --
    auto loadGlobal = [&](int kt) {
        const int k0 = kt * TILE_K;
        #pragma unroll
        for (int i = 0; i < 4; ++i) {
            const int idx = tid + 256 * i;
            const int row = idx >> 3;          // 0..127
            const int c4  = idx & 7;
            ra[i] = *(const float4*)(Sb + (size_t)(c0 + row) * Q_DIM + k0 + c4 * 4);
        }
        #pragma unroll
        for (int i = 0; i < 4; ++i) {
            const int idx = tid + 256 * i;
            const int k   = idx >> 5;          // 0..31
            const int c4  = idx & 31;
            rb[i] = *(const float4*)(Qb + (size_t)(k0 + k) * D_DIM + d0 + c4 * 4);
        }
    };

    // -- drain registers into LDS buffer `buf` (exp transform on A) --
    auto storeLDS = [&](int buf) {
        _Float16* A = &Alds[buf * ABUF];
        _Float16* B = &Blds[buf * BBUF];
        #pragma unroll
        for (int i = 0; i < 4; ++i) {
            const int idx = tid + 256 * i;
            const int row = idx >> 3;
            const int c4  = idx & 7;
            const float mm = rowm[row];
            const float iv = rowinv[row];
            v4h h;
            h.x = (_Float16)(__expf(ra[i].x - mm) * iv);
            h.y = (_Float16)(__expf(ra[i].y - mm) * iv);
            h.z = (_Float16)(__expf(ra[i].z - mm) * iv);
            h.w = (_Float16)(__expf(ra[i].w - mm) * iv);
            *(v4h*)(&A[row * LDA + c4 * 4]) = h;
        }
        #pragma unroll
        for (int i = 0; i < 4; ++i) {
            const int idx = tid + 256 * i;
            const int k   = idx >> 5;
            const int d   = (idx & 31) * 4;
            B[(d + 0) * LDB + k] = (_Float16)rb[i].x;
            B[(d + 1) * LDB + k] = (_Float16)rb[i].y;
            B[(d + 2) * LDB + k] = (_Float16)rb[i].z;
            B[(d + 3) * LDB + k] = (_Float16)rb[i].w;
        }
    };

    loadGlobal(0);
    storeLDS(0);
    __syncthreads();

    for (int kt = 0; kt < K_TILES; ++kt) {
        const int cur = kt & 1;

        // Issue next tile's global loads first (latency hidden by WMMAs).
        if (kt + 1 < K_TILES) loadGlobal(kt + 1);

        // Prefetch tile kt+2 toward L2 (global_prefetch_b8).
        if (kt + 2 < K_TILES) {
            const int k2 = (kt + 2) * TILE_K;
            __builtin_prefetch(Sb + (size_t)(c0 + (tid >> 1)) * Q_DIM + k2, 0, 0);
            __builtin_prefetch(Qb + (size_t)(k2 + (tid >> 3)) * D_DIM + d0 + (tid & 7) * 16, 0, 0);
        }

        // ---- WMMA on current buffer ----
        const _Float16* A = &Alds[cur * ABUF];
        const _Float16* B = &Blds[cur * BBUF];

        // B fragments (16-bit B 32x16, k order used here: lanes 0-15 K=0-15, lanes 16-31
        // K=16-31, N = lane%16) -> contiguous v16h read.
        v16h bfrag[2];
        #pragma unroll
        for (int n = 0; n < 2; ++n) {
            const int bn = nBase + n * 16 + lrow;
            bfrag[n] = *(const v16h*)(&B[bn * LDB + hsel * 16]);
        }

        // A fragments (16-bit A 16x32, k order used here: lanes 0-15 K=0-15,
        // lanes 16-31 K=16-31, M = lane%16) -> two v8h reads.
        #pragma unroll
        for (int m = 0; m < 4; ++m) {
            const int arow = mBase + m * 16 + lrow;
            const _Float16* ap = &A[arow * LDA + hsel * 16];
            v8h lo = *(const v8h*)(ap);
            v8h hi = *(const v8h*)(ap + 8);
            v16h a = __builtin_shufflevector(lo, hi,
                        0, 1, 2, 3, 4, 5, 6, 7, 8, 9, 10, 11, 12, 13, 14, 15);
            #pragma unroll
            for (int n = 0; n < 2; ++n) {
                acc[m][n] = __builtin_amdgcn_wmma_f32_16x16x32_f16(
                    /*neg_a=*/false, a, /*neg_b=*/false, bfrag[n],
                    /*c_mod=*/(short)0, acc[m][n],
                    /*reuse_a=*/false, /*reuse_b=*/false);
            }
        }

        // Drain next tile into the other buffer; nobody reads it this step.
        if (kt + 1 < K_TILES) storeLDS(cur ^ 1);
        __syncthreads();
    }

    // ---------------- Store: D layout -> VGPR r holds M = r + 8*hsel -------
    #pragma unroll
    for (int m = 0; m < 4; ++m) {
        #pragma unroll
        for (int n = 0; n < 2; ++n) {
            const int orow = c0 + mBase + m * 16 + hsel * 8;
            const int ocol = d0 + nBase + n * 16 + lrow;
            float* op = Ob + (size_t)orow * D_DIM + ocol;
            #pragma unroll
            for (int r = 0; r < 8; ++r)
                op[(size_t)r * D_DIM] = acc[m][n][r];
        }
    }
}

extern "C" void kernel_launch(void* const* d_in, const int* in_sizes, int n_in,
                              void* d_out, int out_size, void* d_ws, size_t ws_size,
                              hipStream_t stream) {
    const float* S  = (const float*)d_in[0];   // [32, 2048, 512]
    const float* Qe = (const float*)d_in[1];   // [32, 512, 1024]
    float* out = (float*)d_out;                // [32, 2048, 1024]

    dim3 grid((C_DIM / TILE_C) * (D_DIM / TILE_D), 32);  // (16*8, 32)
    dim3 block(256);
    c2q_attn_wmma_kernel<<<grid, block, 0, stream>>>(S, Qe, out);
}
